// DDI_3367254360364
// MI455X (gfx1250) — matrix operand order repaired and checked by reading, not compiled, not run on hardware
//
#include <hip/hip_runtime.h>
#include <hip/hip_bf16.h>

typedef __attribute__((ext_vector_type(16))) _Float16 v16h;
typedef __attribute__((ext_vector_type(8)))  float    v8f;
typedef unsigned int u32;

#define BATCH 128
#define FDIM  512
#define TDIM  336
#define PATCH 12
#define NW    28          // T / PATCH
#define FT    (FDIM*TDIM)         // 172032
#define COLS  (FDIM*PATCH)        // 6144
#define MROWS (BATCH*PATCH)       // 1536

__device__ __forceinline__ float gelu_exact(float v) {
    return 0.5f * v * (1.0f + erff(v * 0.70710678118654752f));
}

// ---------------- weight conversion: f32 -> f16 (pairs along K are contiguous) --
__global__ __launch_bounds__(256)
void conv_weights(const float* __restrict__ w1, const float* __restrict__ w2,
                  _Float16* __restrict__ w1h, _Float16* __restrict__ w2h) {
    int i = blockIdx.x * 256 + threadIdx.x;   // 1024*256 = 262144 = 512*512
    w1h[i] = (_Float16)w1[i];
    w2h[i] = (_Float16)w2[i];
}

// ---------------- initial BN stats over batch (172032 columns) ------------------
__global__ __launch_bounds__(256)
void bn0_stats(const float* __restrict__ x,
               float* __restrict__ mean0, float* __restrict__ rstd0) {
    int c = blockIdx.x * 256 + threadIdx.x;   // 672*256 = 172032
    float s = 0.f, ss = 0.f;
    for (int b = 0; b < BATCH; ++b) {
        float v = x[b * FT + c];
        s += v; ss += v * v;
    }
    float m = s * (1.0f / BATCH);
    float var = ss * (1.0f / BATCH) - m * m;
    mean0[c] = m;
    rstd0[c] = rsqrtf(fmaxf(var, 0.0f) + 1e-5f);
}

// ---------------- write win0 = xn[:,:,:12] into d_out ---------------------------
__global__ __launch_bounds__(256)
void win0_apply(const float* __restrict__ x,
                const float* __restrict__ mean0, const float* __restrict__ rstd0,
                const float* __restrict__ g0, const float* __restrict__ b0,
                float* __restrict__ out) {
    int e = blockIdx.x * 256 + threadIdx.x;   // 3072*256 = 786432 = B*F*12
    int p = e % PATCH;
    int f = (e / PATCH) & (FDIM - 1);
    int b = e / COLS;
    int c = f * TDIM + p;
    int idx = b * FT + c;
    out[idx] = (x[idx] - mean0[c]) * rstd0[c] * g0[c] + b0[c];
}

// ---------------- per-step BN stats of prev (gather from d_out) -----------------
__global__ __launch_bounds__(256)
void bn_prev_stats(const float* __restrict__ out, int off,
                   float* __restrict__ mean1, float* __restrict__ rstd1) {
    int c = blockIdx.x * 256 + threadIdx.x;   // 24*256 = 6144
    int f = c / PATCH, p = c % PATCH;
    float s = 0.f, ss = 0.f;
    for (int b = 0; b < BATCH; ++b) {
        float v = out[b * FT + f * TDIM + off + p];
        s += v; ss += v * v;
    }
    float m = s * (1.0f / BATCH);
    float var = ss * (1.0f / BATCH) - m * m;
    mean1[c] = m;
    rstd1[c] = rsqrtf(fmaxf(var, 0.0f) + 1e-5f);
}

// ---------------- per-step BN stats of res (flat (B, 6144)) ---------------------
__global__ __launch_bounds__(256)
void bn_res_stats(const float* __restrict__ res,
                  float* __restrict__ mean2, float* __restrict__ rstd2) {
    int c = blockIdx.x * 256 + threadIdx.x;   // 6144
    float s = 0.f, ss = 0.f;
    for (int b = 0; b < BATCH; ++b) {
        float v = res[b * COLS + c];
        s += v; ss += v * v;
    }
    float m = s * (1.0f / BATCH);
    float var = ss * (1.0f / BATCH) - m * m;
    mean2[c] = m;
    rstd2[c] = rsqrtf(fmaxf(var, 0.0f) + 1e-5f);
}

// ---------------- agg: inp = gelu(BN1(prev) @ aggW^T + aggB); res = inp + xw ----
__global__ __launch_bounds__(256)
void agg_step(const float* __restrict__ prevbuf,   // = d_out
              const float* __restrict__ x,
              const float* __restrict__ mean1, const float* __restrict__ rstd1,
              const float* __restrict__ g1, const float* __restrict__ b1,
              const float* __restrict__ aggw, const float* __restrict__ aggb,
              const float* __restrict__ mean0, const float* __restrict__ rstd0,
              const float* __restrict__ g0, const float* __restrict__ b0,
              float* __restrict__ res, int w) {
    __shared__ float sW[PATCH * PATCH];
    __shared__ float sB[PATCH];
    int tid = threadIdx.x;
    if (tid < PATCH * PATCH) sW[tid] = aggw[tid];
    if (tid < PATCH)         sB[tid] = aggb[tid];
    __syncthreads();

    int e = blockIdx.x * 256 + tid;           // 256*256 = 65536 = B*F
    int b = e >> 9, f = e & (FDIM - 1);
    int base = e * TDIM;                      // (b*F+f)*336
    int off = (w - 1) * PATCH;

    float inp[PATCH];
    #pragma unroll
    for (int p = 0; p < PATCH; ++p) {
        int c = f * PATCH + p;
        float pv = prevbuf[base + off + p];
        inp[p] = (pv - mean1[c]) * rstd1[c] * g1[c] + b1[c];
    }
    #pragma unroll
    for (int q = 0; q < PATCH; ++q) {
        float s = sB[q];
        #pragma unroll
        for (int p = 0; p < PATCH; ++p) s += inp[p] * sW[q * PATCH + p];
        s = gelu_exact(s);
        int cx = f * TDIM + w * PATCH + q;
        float xw = (x[b * FT + cx] - mean0[cx]) * rstd0[cx] * g0[cx] + b0[cx];
        res[b * COLS + f * PATCH + q] = s + xw;
    }
}

// ---------------- fused MLP via WMMA: out = res + 0.5*gelu(gelu(t@W1^T)@W2^T) ---
// One workgroup per 16-row m-tile (96 blocks).  8 waves, 4 n-tiles each.
__global__ __launch_bounds__(256)
void mlp_step(const float* __restrict__ res,
              const float* __restrict__ mean2, const float* __restrict__ rstd2,
              const float* __restrict__ g2, const float* __restrict__ b2,
              const _Float16* __restrict__ w1h, const float* __restrict__ b1v,
              const _Float16* __restrict__ w2h, const float* __restrict__ b2v,
              float* __restrict__ out, int w) {
    __shared__ u32 Au[16 * 260];              // A tile: f16 pairs [m][k/2], pad stride
    __shared__ u32 Hu[16 * 264];              // h tile: f16      [m][n],   pad stride

    const int tid = threadIdx.x;
    const int m0 = blockIdx.x * 16;

    // stage A = BN2(res) transposed strip (16 rows x 512 k) as f16 pairs
    #pragma unroll 4
    for (int i = 0; i < 16; ++i) {
        int pi = i * 256 + tid;               // 4096 pairs
        int ml = pi >> 8;
        int kp = pi & 255;
        int m = m0 + ml;
        int b = m / PATCH, p = m % PATCH;
        int f0 = kp * 2;
        int c0 = f0 * PATCH + p;
        int c1 = c0 + PATCH;
        float v0 = (res[b * COLS + c0] - mean2[c0]) * rstd2[c0] * g2[c0] + b2[c0];
        float v1 = (res[b * COLS + c1] - mean2[c1]) * rstd2[c1] * g2[c1] + b2[c1];
        union { u32 u; _Float16 h[2]; } pk;
        pk.h[0] = (_Float16)v0; pk.h[1] = (_Float16)v1;
        Au[ml * 260 + kp] = pk.u;
    }
    __syncthreads();

    const int lane = tid & 31;
    const int wave = tid >> 5;
    const int half = lane >> 4;
    const int lm = lane & 15;
    const u32* w1p = (const u32*)w1h;         // pair (k,k+1) of row n at [n*256 + k/2]
    const u32* w2p = (const u32*)w2h;
    _Float16* Hh = (_Float16*)Hu;

    // GEMM1: h[m, n] = gelu( sum_k A[m,k] * W1[n,k] + b1[n] )
    for (int nt = 0; nt < 4; ++nt) {
        int n0 = (wave * 4 + nt) * 16;
        v8f acc = {};
        for (int k0 = 0; k0 < 512; k0 += 32) {
            union { u32 u[8]; v16h v; } af, bf;
            #pragma unroll
            for (int v = 0; v < 8; ++v) {
                int ka = k0 + (v >> 2) * 16 + half * 8 + (v & 3) * 2;
                af.u[v] = Au[lm * 260 + (ka >> 1)];
                int kb = k0 + (v >> 2) * 16 + half * 8 + (v & 3) * 2;
                bf.u[v] = w1p[(n0 + lm) * 256 + (kb >> 1)];
            }
            acc = __builtin_amdgcn_wmma_f32_16x16x32_f16(
                false, af.v, false, bf.v, (short)0, acc, false, false);
        }
        int n = n0 + lm;
        float bias = b1v[n];
        #pragma unroll
        for (int v = 0; v < 8; ++v) {
            int ml = v + half * 8;
            Hh[ml * 528 + n] = (_Float16)gelu_exact(acc[v] + bias);
        }
    }
    __syncthreads();

    // GEMM2: out[m, f] = res + 0.5 * gelu( sum_k H[m,k] * W2[f,k] + b2[f] )
    for (int nt = 0; nt < 4; ++nt) {
        int n0 = (wave * 4 + nt) * 16;
        v8f acc = {};
        for (int k0 = 0; k0 < 512; k0 += 32) {
            union { u32 u[8]; v16h v; } af, bf;
            #pragma unroll
            for (int v = 0; v < 8; ++v) {
                int ka = k0 + (v >> 2) * 16 + half * 8 + (v & 3) * 2;
                af.u[v] = Hu[lm * 264 + (ka >> 1)];
                int kb = k0 + (v >> 2) * 16 + half * 8 + (v & 3) * 2;
                bf.u[v] = w2p[(n0 + lm) * 256 + (kb >> 1)];
            }
            acc = __builtin_amdgcn_wmma_f32_16x16x32_f16(
                false, af.v, false, bf.v, (short)0, acc, false, false);
        }
        int f = n0 + lm;
        float bias = b2v[f];
        #pragma unroll
        for (int v = 0; v < 8; ++v) {
            int m = m0 + v + half * 8;
            int b = m / PATCH, p = m % PATCH;
            float hv = gelu_exact(acc[v] + bias);
            out[b * FT + f * TDIM + w * PATCH + p] =
                res[b * COLS + f * PATCH + p] + 0.5f * hv;
        }
    }
}

extern "C" void kernel_launch(void* const* d_in, const int* in_sizes, int n_in,
                              void* d_out, int out_size, void* d_ws, size_t ws_size,
                              hipStream_t stream) {
    const float* x    = (const float*)d_in[0];
    const float* ng   = (const float*)d_in[1];
    const float* nb   = (const float*)d_in[2];
    const float* n1g  = (const float*)d_in[3];
    const float* n1b  = (const float*)d_in[4];
    const float* n2g  = (const float*)d_in[5];
    const float* n2b  = (const float*)d_in[6];
    const float* aggw = (const float*)d_in[7];
    const float* aggb = (const float*)d_in[8];
    const float* fc1w = (const float*)d_in[9];
    const float* fc1b = (const float*)d_in[10];
    const float* fc2w = (const float*)d_in[11];
    const float* fc2b = (const float*)d_in[12];
    float* out = (float*)d_out;

    // workspace layout (floats)
    float* wsf   = (float*)d_ws;
    float* mean0 = wsf;                       // 172032
    float* rstd0 = mean0 + FT;                // 172032
    float* mean1 = rstd0 + FT;                // 6144
    float* rstd1 = mean1 + COLS;
    float* mean2 = rstd1 + COLS;
    float* rstd2 = mean2 + COLS;
    float* resb  = rstd2 + COLS;              // 786432
    _Float16* w1h = (_Float16*)(resb + BATCH * COLS);   // 262144 halves
    _Float16* w2h = w1h + FDIM * FDIM;

    conv_weights<<<1024, 256, 0, stream>>>(fc1w, fc2w, w1h, w2h);
    bn0_stats<<<FT / 256, 256, 0, stream>>>(x, mean0, rstd0);
    win0_apply<<<(BATCH * COLS) / 256, 256, 0, stream>>>(x, mean0, rstd0, ng, nb, out);

    for (int w = 1; w < NW; ++w) {
        bn_prev_stats<<<COLS / 256, 256, 0, stream>>>(out, (w - 1) * PATCH, mean1, rstd1);
        agg_step<<<(BATCH * FDIM) / 256, 256, 0, stream>>>(
            out, x, mean1, rstd1, n1g, n1b, aggw, aggb, mean0, rstd0, ng, nb, resb, w);
        bn_res_stats<<<COLS / 256, 256, 0, stream>>>(resb, mean2, rstd2);
        mlp_step<<<MROWS / 16, 256, 0, stream>>>(
            resb, mean2, rstd2, n2g, n2b, w1h, fc1b, w2h, fc2b, out, w);
    }
}
